// GCN3LayerConcat_26560077758927
// MI455X (gfx1250) — matrix operand order repaired and checked by reading, not compiled, not run on hardware
//
#include <hip/hip_runtime.h>
#include <hip/hip_bf16.h>
#include <math.h>

// ---------------------------------------------------------------------------
// GCN 3-layer + concat-FC for MI455X (gfx1250, wave32, WMMA bf16 -> f32)
// ---------------------------------------------------------------------------

typedef __bf16 bf16_t;
typedef __bf16 v16bf __attribute__((ext_vector_type(16)));
typedef __bf16 v8bf  __attribute__((ext_vector_type(8)));
typedef __bf16 v4bf  __attribute__((ext_vector_type(4)));
typedef float  v8f   __attribute__((ext_vector_type(8)));

#define N_NODES   64000
#define N_EDGES   1048576
#define NB_NODES  1000
#define N_GRAPHS  64
#define F_IN      128
#define H1        256
#define H2        256
#define H3        128
#define OUT_F     64
#define FC_K      (NB_NODES * H3)   // 128000

static inline int cdiv(long a, long b) { return (int)((a + b - 1) / b); }

__device__ __forceinline__ v8f wmma_bf16(v16bf a, v16bf b, v8f c) {
    return __builtin_amdgcn_wmma_f32_16x16x32_bf16(
        false, a, false, b, (short)0, c, false, false);
}

__device__ __forceinline__ v16bf load_a_frag(const bf16_t* arowp, int k0, int g) {
    // A 16x32 fragment: K=k0+16g+j
    return *(const v16bf*)(arowp + k0 + 16 * g);
}

// ---------------------------------------------------------------------------
// small utility kernels
// ---------------------------------------------------------------------------
__global__ void k_fill_f32(float* p, float v, int n) {
    int i = blockIdx.x * blockDim.x + threadIdx.x;
    if (i < n) p[i] = v;
}

__global__ void k_deg_accum(const int* __restrict__ dste, float* __restrict__ deg, int e) {
    int i = blockIdx.x * blockDim.x + threadIdx.x;
    if (i < e) atomicAdd(&deg[dste[i]], 1.0f);
}

__global__ void k_rsqrt_inplace(float* p, int n) {
    int i = blockIdx.x * blockDim.x + threadIdx.x;
    if (i < n) p[i] = rsqrtf(p[i]);
}

__global__ void k_f32_to_bf16(const float* __restrict__ s, bf16_t* __restrict__ d, long n) {
    long stride = (long)gridDim.x * blockDim.x;
    for (long i = blockIdx.x * (long)blockDim.x + threadIdx.x; i < n; i += stride)
        d[i] = (bf16_t)s[i];
}

// ---------------------------------------------------------------------------
// Weight pre-pack: row-major f32 W[K,Nc] -> bf16 B-fragment order so each
// lane's 16-element WMMA B operand is one contiguous 32-byte load.
//   Bpack[(((kc*tilesN)+tn)*32 + lane)*16 + j] = W[kc*32 + j + 16*(lane/16), tn*16 + lane%16]
// ---------------------------------------------------------------------------
__global__ void k_pack_B(const float* __restrict__ W, bf16_t* __restrict__ P,
                         int K, int Nc) {
    int tilesN = Nc >> 4;
    long total = (long)K * Nc;
    long stride = (long)gridDim.x * blockDim.x;
    for (long i = blockIdx.x * (long)blockDim.x + threadIdx.x; i < total; i += stride) {
        int  j    = (int)(i & 15);
        int  lane = (int)((i >> 4) & 31);
        long frag = i >> 9;                     // kc*tilesN + tn
        int  tn   = (int)(frag % tilesN);
        int  kc   = (int)(frag / tilesN);
        int  krow = (kc << 5) + j + ((lane >> 4) << 4);
        int  col  = (tn << 4) + (lane & 15);
        P[i] = (bf16_t)W[(long)krow * Nc + col];
    }
}

// ---------------------------------------------------------------------------
// Dense GEMM: C[M,Nc](f32) = A[M,K](bf16 row major) x Bpack (fragment order)
// One wave computes a 16x64 output band: 1 A fragment feeds 4 WMMAs per step.
// K is a template parameter -> fully unrolled K loop, deep load pipelining.
// __launch_bounds__(256,1): allow high VGPR count so B fragments of one step
// live in distinct registers (staggered s_wait_loadcnt instead of 0x0 stalls).
// ---------------------------------------------------------------------------
template <int K>
__global__ __launch_bounds__(256, 1)
void k_gemm_bf16_wmma(const bf16_t* __restrict__ A,
                      const bf16_t* __restrict__ Bp,
                      float* __restrict__ C,
                      int M, int Nc) {
    int wave = (int)((blockIdx.x * (long)blockDim.x + threadIdx.x) >> 5);
    int lane = threadIdx.x & 31;
    int tilesN  = Nc >> 4;
    int ngroups = tilesN >> 2;                  // 4 N-tiles per wave
    int groups  = (M >> 4) * ngroups;
    if (wave >= groups) return;                 // wave-uniform: EXEC all-ones

    int tm = wave / ngroups;
    int tg = wave - tm * ngroups;               // tile group: tn0 = 4*tg
    int g   = lane >> 4;
    int l16 = lane & 15;
    const bf16_t* arowp = A + (long)(tm * 16 + l16) * K;

    v8f c0, c1, c2, c3;
#pragma unroll
    for (int r = 0; r < 8; ++r) { c0[r] = 0.f; c1[r] = 0.f; c2[r] = 0.f; c3[r] = 0.f; }

    constexpr int NK = K >> 5;
#pragma unroll
    for (int kc = 0; kc < NK; ++kc) {
        int k0 = kc << 5;
        __builtin_prefetch(arowp + k0 + 64, 0, 1);   // global_prefetch_b8

        v16bf a = load_a_frag(arowp, k0, g);

        const bf16_t* bbase =
            Bp + ((((long)kc * tilesN + (tg << 2)) * 32 + lane) << 4);
        v16bf b0 = *(const v16bf*)(bbase);
        v16bf b1 = *(const v16bf*)(bbase + 512);
        v16bf b2 = *(const v16bf*)(bbase + 1024);
        v16bf b3 = *(const v16bf*)(bbase + 1536);

        c0 = wmma_bf16(a, b0, c0);
        c1 = wmma_bf16(a, b1, c1);
        c2 = wmma_bf16(a, b2, c2);
        c3 = wmma_bf16(a, b3, c3);
    }

    // C layout: VGPR r -> row rbase + r, col = tile*16 + lane%16
    int rbase = tm * 16 + (g << 3);
    int cbase = (tg << 2) * 16 + l16;
#pragma unroll
    for (int r = 0; r < 8; ++r) {
        long row = (long)(rbase + r) * Nc;
        C[row + cbase]      = c0[r];
        C[row + cbase + 16] = c1[r];
        C[row + cbase + 32] = c2[r];
        C[row + cbase + 48] = c3[r];
    }
}

// ---------------------------------------------------------------------------
// Aggregation: agg = D^-1/2 (A+I) D^-1/2 * hl   (float4-wide, atomic scatter)
// ---------------------------------------------------------------------------
__global__ void k_self_loop(const float* __restrict__ hl,
                            const float* __restrict__ dinv,
                            float* __restrict__ agg, int H) {
    int hq = H >> 2;
    long total = (long)N_NODES * hq;
    long stride = (long)gridDim.x * blockDim.x;
    for (long i = blockIdx.x * (long)blockDim.x + threadIdx.x; i < total; i += stride) {
        int node = (int)(i / hq);
        float w = dinv[node];
        float4 v = ((const float4*)hl)[i];
        float s = w * w;
        v.x *= s; v.y *= s; v.z *= s; v.w *= s;
        ((float4*)agg)[i] = v;
    }
}

__global__ void k_edge_scatter(const int* __restrict__ srce,
                               const int* __restrict__ dste,
                               const float* __restrict__ dinv,
                               const float* __restrict__ hl,
                               float* __restrict__ agg, int H) {
    int hq = H >> 2;
    long total = (long)N_EDGES * hq;
    long stride = (long)gridDim.x * blockDim.x;
    for (long i = blockIdx.x * (long)blockDim.x + threadIdx.x; i < total; i += stride) {
        int e  = (int)(i / hq);
        int f4 = (int)(i - (long)e * hq);
        int s = srce[e];
        int d = dste[e];
        float w = dinv[s] * dinv[d];
        float4 v = ((const float4*)(hl + (long)s * H))[f4];
        float* dp = agg + (long)d * H + (f4 << 2);
        atomicAdd(dp + 0, v.x * w);
        atomicAdd(dp + 1, v.y * w);
        atomicAdd(dp + 2, v.z * w);
        atomicAdd(dp + 3, v.w * w);
    }
}

__global__ void k_bias_tanh_bf16(const float* __restrict__ agg,
                                 const float* __restrict__ bias,
                                 bf16_t* __restrict__ outb, int H) {
    int hq = H >> 2;
    long total = (long)N_NODES * hq;
    long stride = (long)gridDim.x * blockDim.x;
    for (long i = blockIdx.x * (long)blockDim.x + threadIdx.x; i < total; i += stride) {
        int f4 = (int)(i % hq);
        float4 v = ((const float4*)agg)[i];
        const float* bp = bias + (f4 << 2);
        v4bf o;
        o[0] = (bf16_t)tanhf(v.x + bp[0]);
        o[1] = (bf16_t)tanhf(v.y + bp[1]);
        o[2] = (bf16_t)tanhf(v.z + bp[2]);
        o[3] = (bf16_t)tanhf(v.w + bp[3]);
        ((v4bf*)outb)[i] = o;
    }
}

// ---------------------------------------------------------------------------
// Final FC: out[64,64] = Hc[64, 128000] @ Wfc[128000, 64] + bfc
// Split-K WMMA (wave = 16x64 band x K-chunk), f32 atomic reduction.
// KCHUNK is compile-time -> fully unrolled inner loop.
// ---------------------------------------------------------------------------
__global__ void k_fc_init(float* __restrict__ out, const float* __restrict__ bfc) {
    int i = blockIdx.x * blockDim.x + threadIdx.x;
    if (i < N_GRAPHS * OUT_F) out[i] = bfc[i & (OUT_F - 1)];
}

template <int KCHUNK>
__global__ __launch_bounds__(256, 1)
void k_fc_wmma_splitk(const bf16_t* __restrict__ Hc,
                      const bf16_t* __restrict__ Wp,
                      float* __restrict__ out,
                      int nsplits) {
    int wave = (int)((blockIdx.x * (long)blockDim.x + threadIdx.x) >> 5);
    int lane = threadIdx.x & 31;
    if (wave >= 4 * nsplits) return;            // wave-uniform

    int tm = wave & 3;                          // 4 row tiles (64 rows)
    int s  = wave >> 2;                         // K split
    int g   = lane >> 4;
    int l16 = lane & 15;
    const bf16_t* arowp = Hc + (long)(tm * 16 + l16) * FC_K;
    const int tilesN = OUT_F / 16;              // 4 -> one group covers all cols

    v8f c0, c1, c2, c3;
#pragma unroll
    for (int r = 0; r < 8; ++r) { c0[r] = 0.f; c1[r] = 0.f; c2[r] = 0.f; c3[r] = 0.f; }

    int kc0 = (s * KCHUNK) >> 5;
    constexpr int NK = KCHUNK >> 5;
#pragma unroll
    for (int kk = 0; kk < NK; ++kk) {
        int kc = kc0 + kk;
        int k0 = kc << 5;
        __builtin_prefetch(arowp + k0 + 64, 0, 1);

        v16bf a = load_a_frag(arowp, k0, g);

        const bf16_t* bbase = Wp + ((((long)kc * tilesN) * 32 + lane) << 4);
        v16bf b0 = *(const v16bf*)(bbase);
        v16bf b1 = *(const v16bf*)(bbase + 512);
        v16bf b2 = *(const v16bf*)(bbase + 1024);
        v16bf b3 = *(const v16bf*)(bbase + 1536);

        c0 = wmma_bf16(a, b0, c0);
        c1 = wmma_bf16(a, b1, c1);
        c2 = wmma_bf16(a, b2, c2);
        c3 = wmma_bf16(a, b3, c3);
    }

    int rbase = tm * 16 + (g << 3);
#pragma unroll
    for (int r = 0; r < 8; ++r) {
        float* rowp = out + (rbase + r) * OUT_F + l16;
        atomicAdd(rowp + 0,  c0[r]);
        atomicAdd(rowp + 16, c1[r]);
        atomicAdd(rowp + 32, c2[r]);
        atomicAdd(rowp + 48, c3[r]);
    }
}

// ---------------------------------------------------------------------------
// host-side orchestration
// ---------------------------------------------------------------------------
extern "C" void kernel_launch(void* const* d_in, const int* in_sizes, int n_in,
                              void* d_out, int out_size, void* d_ws, size_t ws_size,
                              hipStream_t stream) {
    const float* x   = (const float*)d_in[0];
    const int*   ei  = (const int*)d_in[1];   // [2, E]
    const float* W1  = (const float*)d_in[3];
    const float* b1  = (const float*)d_in[4];
    const float* W2  = (const float*)d_in[5];
    const float* b2  = (const float*)d_in[6];
    const float* W3  = (const float*)d_in[7];
    const float* b3  = (const float*)d_in[8];
    const float* Wfc = (const float*)d_in[9];
    const float* bfc = (const float*)d_in[10];
    float* out = (float*)d_out;

    const int* srcE = ei;
    const int* dstE = ei + N_EDGES;

    // workspace carve-out
    char* ws = (char*)d_ws;
    size_t off = 0;
    auto carve = [&](size_t bytes) -> void* {
        void* p = ws + off;
        off += (bytes + 255) & ~(size_t)255;
        return p;
    };
    float*  dinv = (float*)carve((size_t)N_NODES * 4);
    float*  hl   = (float*)carve((size_t)N_NODES * 256 * 4);
    float*  agg  = (float*)carve((size_t)N_NODES * 256 * 4);
    bf16_t* hb   = (bf16_t*)carve((size_t)N_NODES * 256 * 2);
    bf16_t* wb   = (bf16_t*)carve((size_t)FC_K * OUT_F * 2);

    const int T = 256;                 // 8 waves / block
    const int GS_CAP = 1 << 18;        // grid-stride cap

    // --- symmetric normalization: deg -> dinv ---
    k_fill_f32<<<cdiv(N_NODES, T), T, 0, stream>>>(dinv, 1.0f, N_NODES);
    k_deg_accum<<<cdiv(N_EDGES, T), T, 0, stream>>>(dstE, dinv, N_EDGES);
    k_rsqrt_inplace<<<cdiv(N_NODES, T), T, 0, stream>>>(dinv, N_NODES);

    // --- layer 1: x(128) -> h(256) ---
    {
        long nx = (long)N_NODES * F_IN;
        k_f32_to_bf16<<<min(cdiv(nx, T), GS_CAP), T, 0, stream>>>(x, hb, nx);
        long nw = (long)F_IN * H1;
        k_pack_B<<<min(cdiv(nw, T), GS_CAP), T, 0, stream>>>(W1, wb, F_IN, H1);
        int groups = (N_NODES / 16) * (H1 / 64);
        k_gemm_bf16_wmma<F_IN><<<cdiv(groups, 8), T, 0, stream>>>(hb, wb, hl, N_NODES, H1);
        long na4 = (long)N_NODES * H1 / 4;
        k_self_loop<<<min(cdiv(na4, T), GS_CAP), T, 0, stream>>>(hl, dinv, agg, H1);
        long ne4 = (long)N_EDGES * H1 / 4;
        k_edge_scatter<<<min(cdiv(ne4, T), GS_CAP), T, 0, stream>>>(srcE, dstE, dinv, hl, agg, H1);
        k_bias_tanh_bf16<<<min(cdiv(na4, T), GS_CAP), T, 0, stream>>>(agg, b1, hb, H1);
    }

    // --- layer 2: h(256) -> h(256) ---
    {
        long nw = (long)H1 * H2;
        k_pack_B<<<min(cdiv(nw, T), GS_CAP), T, 0, stream>>>(W2, wb, H1, H2);
        int groups = (N_NODES / 16) * (H2 / 64);
        k_gemm_bf16_wmma<H1><<<cdiv(groups, 8), T, 0, stream>>>(hb, wb, hl, N_NODES, H2);
        long na4 = (long)N_NODES * H2 / 4;
        k_self_loop<<<min(cdiv(na4, T), GS_CAP), T, 0, stream>>>(hl, dinv, agg, H2);
        long ne4 = (long)N_EDGES * H2 / 4;
        k_edge_scatter<<<min(cdiv(ne4, T), GS_CAP), T, 0, stream>>>(srcE, dstE, dinv, hl, agg, H2);
        k_bias_tanh_bf16<<<min(cdiv(na4, T), GS_CAP), T, 0, stream>>>(agg, b2, hb, H2);
    }

    // --- layer 3: h(256) -> h(128) ---
    {
        long nw = (long)H2 * H3;
        k_pack_B<<<min(cdiv(nw, T), GS_CAP), T, 0, stream>>>(W3, wb, H2, H3);
        int groups = (N_NODES / 16) * (H3 / 64);
        k_gemm_bf16_wmma<H2><<<cdiv(groups, 8), T, 0, stream>>>(hb, wb, hl, N_NODES, H3);
        long na4 = (long)N_NODES * H3 / 4;
        k_self_loop<<<min(cdiv(na4, T), GS_CAP), T, 0, stream>>>(hl, dinv, agg, H3);
        long ne4 = (long)N_EDGES * H3 / 4;
        k_edge_scatter<<<min(cdiv(ne4, T), GS_CAP), T, 0, stream>>>(srcE, dstE, dinv, hl, agg, H3);
        k_bias_tanh_bf16<<<min(cdiv(na4, T), GS_CAP), T, 0, stream>>>(agg, b3, hb, H3);
    }

    // --- final FC: [64,128000] @ [128000,64] + bfc (split-K) ---
    {
        long nw = (long)FC_K * OUT_F;
        k_pack_B<<<min(cdiv(nw, T), GS_CAP), T, 0, stream>>>(Wfc, wb, FC_K, OUT_F);
        k_fc_init<<<cdiv(N_GRAPHS * OUT_F, T), T, 0, stream>>>(out, bfc);
        constexpr int KCHUNK  = 256;           // 8 WMMA steps per split
        const int nsplits = FC_K / KCHUNK;     // 500
        int waves = 4 * nsplits;               // 2000 waves
        k_fc_wmma_splitk<KCHUNK><<<cdiv(waves, 8), T, 0, stream>>>(hb, wb, out, nsplits);
    }
}
